// DLRM_72636486910486
// MI455X (gfx1250) — matrix operand order repaired and checked by reading, not compiled, not run on hardware
//
#include <hip/hip_runtime.h>
#include <hip/hip_bf16.h>

typedef __attribute__((ext_vector_type(16))) __bf16 v16bf;
typedef __attribute__((ext_vector_type(8)))  float  v8f;

#define F_FEAT 26
#define VOC    100000
#define DIMD   64
#define BATCH  16384
#define NPAIR  325                     // F*(F-1)/2
#define INDIM  (F_FEAT*DIMD + NPAIR)   // 1989
#define KPAD   2048
#define H1N    512
#define H2N    256

// ---------------------------------------------------------------------------
// Kernel 1: embedding gather + pairwise interaction + bf16 pack of joint row
// One block (4 waves) per sample. Gather is the HBM-bound part of the whole
// pipeline (~109MB of random 256B rows); everything is float4 / coalesced.
// ---------------------------------------------------------------------------
__global__ __launch_bounds__(128) void dlrm_gather_interact(
    const int* __restrict__ indices, const float* __restrict__ emb,
    __bf16* __restrict__ joint)
{
    __shared__ float Ef[F_FEAT][DIMD + 4];   // +4 pad: dodge 64-bank conflicts
    __shared__ int sIdx[F_FEAT];
    const int b   = blockIdx.x;
    const int tid = threadIdx.x;

    if (tid < F_FEAT) sIdx[tid] = indices[tid * BATCH + b];
    __syncthreads();

    // gather: 26 rows x 16 float4
    for (int q = tid; q < F_FEAT * (DIMD / 4); q += 128) {
        const int f  = q >> 4;
        const int c4 = q & 15;
        const float* src = emb + ((size_t)f * VOC + (size_t)sIdx[f]) * DIMD + c4 * 4;
        float4 v = *(const float4*)src;
        Ef[f][c4 * 4 + 0] = v.x; Ef[f][c4 * 4 + 1] = v.y;
        Ef[f][c4 * 4 + 2] = v.z; Ef[f][c4 * 4 + 3] = v.w;
    }
    __syncthreads();

    __bf16* row = joint + (size_t)b * KPAD;

    // dense part of joint: E flattened (F*D = 1664)
    for (int q = tid; q < F_FEAT * DIMD; q += 128)
        row[q] = (__bf16)Ef[q >> 6][q & 63];

    // interaction: triu(E E^T, k=1), row-major pair order (= jnp.triu_indices)
    for (int p = tid; p < NPAIR; p += 128) {
        int i = 0, rem = p, cnt = F_FEAT - 1;
        while (rem >= cnt) { rem -= cnt; ++i; --cnt; }
        const int j = i + 1 + rem;
        float s = 0.f;
        #pragma unroll
        for (int d = 0; d < DIMD; ++d) s += Ef[i][d] * Ef[j][d];
        row[F_FEAT * DIMD + p] = (__bf16)s;
    }

    // zero pad 1989..2047 so GEMM K=2048 is exact
    for (int c = INDIM + tid; c < KPAD; c += 128)
        row[c] = (__bf16)0.f;
}

// ---------------------------------------------------------------------------
// Kernel 2: weights fp32 -> bf16, TRANSPOSED to N-major (K-contiguous) so the
// GEMM's B-tile staging is a plain 128-bit copy with no in-loop transpose.
// W1t: [512][2048] (K cols 1989..2047 zero-padded), W2t: [256][512].
// One-time 4MB kernel; strided fp32 reads are irrelevant at this size.
// ---------------------------------------------------------------------------
__global__ void dlrm_prep_weights(const float* __restrict__ W1,
                                  const float* __restrict__ W2,
                                  __bf16* __restrict__ W1t,
                                  __bf16* __restrict__ W2t)
{
    const int stride = gridDim.x * blockDim.x;
    const int gid = blockIdx.x * blockDim.x + threadIdx.x;
    for (int i = gid; i < H1N * KPAD; i += stride) {
        const int n = i >> 11;           // / 2048
        const int k = i & (KPAD - 1);
        W1t[i] = (k < INDIM) ? (__bf16)W1[(size_t)k * H1N + n] : (__bf16)0.f;
    }
    for (int i = gid; i < H2N * H1N; i += stride) {
        const int n = i >> 9;            // / 512
        const int k = i & (H1N - 1);
        W2t[i] = (__bf16)W2[(size_t)k * H2N + n];
    }
}

// ---------------------------------------------------------------------------
// WMMA bf16 GEMM + bias + ReLU, C in bf16.  A: MxK row-major. Bw: NxK
// (pre-transposed weights), so both LDS tiles stage as pure b128 copies.
// Block tile 128x128, 8 waves as 2(M) x 4(N); each wave computes 64x32 =
// 8 x v_wmma_f32_16x16x32_bf16 per K-step of 32 against 6 fragment loads,
// each fragment = two contiguous ds_load_b128 (ISA 7.12.2 layouts).
// Ping-pong LDS buffers; the next tile's 4 global b128 loads live in SCALAR
// uint4s manipulated by macros (no lambda-captured arrays -> nothing for
// SROA to miss -> no scratch alloca in the hot loop).
// ---------------------------------------------------------------------------
template<int KTOT, int NTOT>
__global__ __attribute__((amdgpu_flat_work_group_size(256, 256),
                          amdgpu_waves_per_eu(1)))
void dlrm_gemm_bias_relu(
    const __bf16* __restrict__ A, const __bf16* __restrict__ Bw,
    const float* __restrict__ bias, __bf16* __restrict__ C)
{
    constexpr int BM = 128, BN = 128, KT = 32;
    constexpr int AST = KT + 8;    // 40 elems = 80B row stride (16B aligned)
    __shared__ __align__(16) __bf16 As[2][BM][AST];   // [buf][m][k]
    __shared__ __align__(16) __bf16 Bt[2][BN][AST];   // [buf][n][k]

    const int tid   = threadIdx.x;
    const int lane  = tid & 31;
    const int wid   = tid >> 5;
    const int waveM = wid & 1;          // 2 waves over M, 64 rows each
    const int waveN = wid >> 1;         // 4 waves over N, 32 cols each
    const int rowBase = blockIdx.x * BM;
    const int colBase = blockIdx.y * BN;

    v8f acc[4][2];
    #pragma unroll
    for (int mt = 0; mt < 4; ++mt)
        #pragma unroll
        for (int nt = 0; nt < 2; ++nt)
            #pragma unroll
            for (int e = 0; e < 8; ++e) acc[mt][nt][e] = 0.f;

    // per-lane fragment coordinates (ISA 7.12.2, 16-bit operands, wave32)
    const int arow  = lane & 15;            // A: M row within 16
    const int akoff = (lane >> 4) << 3;     // A: lanes 16-31 hold K groups +8
    const int bn    = lane & 15;            // B: N column within 16
    const int bkoff = (lane >> 4) << 3;     // B: lanes 16-31 hold K groups +8

    // staging: both tiles are 128 rows x 32 K (512 uint4, 2/thread each)
    const int r0 = tid >> 2, r1 = (tid >> 2) + 64, g = tid & 3;
    const __bf16* aPtr0 = A  + (size_t)(rowBase + r0) * KTOT + g * 8;
    const __bf16* aPtr1 = A  + (size_t)(rowBase + r1) * KTOT + g * 8;
    const __bf16* bPtr0 = Bw + (size_t)(colBase + r0) * KTOT + g * 8;
    const __bf16* bPtr1 = Bw + (size_t)(colBase + r1) * KTOT + g * 8;

    uint4 a0, a1, b0, b1;   // scalar prefetch registers (never address-taken)

#define DLRM_LOADG(k0_)                                 \
    do {                                                \
        a0 = *(const uint4*)(aPtr0 + (k0_));            \
        a1 = *(const uint4*)(aPtr1 + (k0_));            \
        b0 = *(const uint4*)(bPtr0 + (k0_));            \
        b1 = *(const uint4*)(bPtr1 + (k0_));            \
    } while (0)

#define DLRM_STORELDS(buf_)                             \
    do {                                                \
        *(uint4*)&As[buf_][r0][g * 8] = a0;             \
        *(uint4*)&As[buf_][r1][g * 8] = a1;             \
        *(uint4*)&Bt[buf_][r0][g * 8] = b0;             \
        *(uint4*)&Bt[buf_][r1][g * 8] = b1;             \
    } while (0)

    auto compute = [&](int buf) {
        v16bf afrag[4], bfrag[2];
        #pragma unroll
        for (int mt = 0; mt < 4; ++mt) {
            const int m = waveM * 64 + mt * 16 + arow;
            union { uint4 q[2]; v16bf v; } u;
            u.q[0] = *(const uint4*)&As[buf][m][akoff];        // K base..base+7
            u.q[1] = *(const uint4*)&As[buf][m][akoff + 16];   // K base+16..+23
            afrag[mt] = u.v;
        }
        #pragma unroll
        for (int nt = 0; nt < 2; ++nt) {
            const int n = waveN * 32 + nt * 16 + bn;
            union { uint4 q[2]; v16bf v; } u;
            u.q[0] = *(const uint4*)&Bt[buf][n][bkoff];        // K bkoff..+7
            u.q[1] = *(const uint4*)&Bt[buf][n][bkoff + 16];   // K bkoff+16..+23
            bfrag[nt] = u.v;
        }
        #pragma unroll
        for (int mt = 0; mt < 4; ++mt)
            #pragma unroll
            for (int nt = 0; nt < 2; ++nt)
                acc[mt][nt] = __builtin_amdgcn_wmma_f32_16x16x32_bf16(
                    false, afrag[mt], false, bfrag[nt],
                    (short)0, acc[mt][nt], false, false);
    };

    const int nK = KTOT / KT;
    DLRM_LOADG(0);
    DLRM_STORELDS(0);
    __syncthreads();
    for (int kk = 0; kk < nK; ++kk) {
        const int cur = kk & 1;
        if (kk + 1 < nK) DLRM_LOADG((kk + 1) * KT);   // overlap with compute
        if (kk + 2 < nK) {                            // -> global_prefetch_b8
            __builtin_prefetch(aPtr0 + (kk + 2) * KT, 0, 3);
            __builtin_prefetch(bPtr0 + (kk + 2) * KT, 0, 3);
        }
        compute(cur);
        if (kk + 1 < nK) DLRM_STORELDS(cur ^ 1);      // other buffer: safe
        __syncthreads();
    }
#undef DLRM_LOADG
#undef DLRM_STORELDS

    // epilogue: C layout (f32 16x16): M = vgpr + 8*(lane>=16), N = lane&15
    const int cn  = lane & 15;
    const int cmo = (lane >> 4) << 3;
    #pragma unroll
    for (int mt = 0; mt < 4; ++mt)
        #pragma unroll
        for (int nt = 0; nt < 2; ++nt) {
            const int col = colBase + waveN * 32 + nt * 16 + cn;
            const float bv = bias[col];
            #pragma unroll
            for (int e = 0; e < 8; ++e) {
                const int r = rowBase + waveM * 64 + mt * 16 + cmo + e;
                float v = acc[mt][nt][e] + bv;
                v = v > 0.f ? v : 0.f;
                C[(size_t)r * NTOT + col] = (__bf16)v;
            }
        }
}

// ---------------------------------------------------------------------------
// Kernel 5: out[b] = h2[b,:] . W3 + b3   (one wave32 per row)
// ---------------------------------------------------------------------------
__global__ __launch_bounds__(256) void dlrm_final(
    const __bf16* __restrict__ h2, const float* __restrict__ W3,
    const float* __restrict__ b3, float* __restrict__ out)
{
    const int row  = blockIdx.x * 8 + (threadIdx.x >> 5);
    const int lane = threadIdx.x & 31;
    float s = 0.f;
    #pragma unroll
    for (int c = lane; c < H2N; c += 32)
        s += (float)h2[(size_t)row * H2N + c] * W3[c];
    #pragma unroll
    for (int off = 16; off > 0; off >>= 1)
        s += __shfl_down(s, off, 32);
    if (lane == 0) out[row] = s + b3[0];
}

// ---------------------------------------------------------------------------
extern "C" void kernel_launch(void* const* d_in, const int* in_sizes, int n_in,
                              void* d_out, int out_size, void* d_ws, size_t ws_size,
                              hipStream_t stream) {
    (void)in_sizes; (void)n_in; (void)out_size; (void)ws_size;
    const int*   indices = (const int*)  d_in[0];
    const float* emb     = (const float*)d_in[1];
    const float* W1      = (const float*)d_in[2];
    const float* b1      = (const float*)d_in[3];
    const float* W2      = (const float*)d_in[4];
    const float* b2      = (const float*)d_in[5];
    const float* W3      = (const float*)d_in[6];
    const float* b3      = (const float*)d_in[7];
    float* out = (float*)d_out;

    char* ws = (char*)d_ws;
    // workspace layout (bytes), all 16B aligned; total ~90.3 MB
    // (joint 64MB + h1 16MB + h2 8MB = 88MB of intermediates < 192MB L2)
    __bf16* joint = (__bf16*)(ws);                               // 16384*2048*2 = 64 MB
    __bf16* W1t   = (__bf16*)(ws + ((size_t)64 << 20));          // 512*2048*2   =  2 MB
    __bf16* h1    = (__bf16*)(ws + ((size_t)66 << 20));          // 16384*512*2  = 16 MB
    __bf16* W2t   = (__bf16*)(ws + ((size_t)82 << 20));          // 256*512*2    = .25MB
    __bf16* h2    = (__bf16*)(ws + ((size_t)83 << 20));          // 16384*256*2  =  8 MB

    dlrm_prep_weights<<<256, 256, 0, stream>>>(W1, W2, W1t, W2t);
    dlrm_gather_interact<<<BATCH, 128, 0, stream>>>(indices, emb, joint);
    dlrm_gemm_bias_relu<KPAD, H1N><<<dim3(BATCH / 128, H1N / 128), 256, 0, stream>>>(joint, W1t, b1, h1);
    dlrm_gemm_bias_relu<H1N, H2N><<<dim3(BATCH / 128, H2N / 128), 256, 0, stream>>>(h1,    W2t, b2, h2);
    dlrm_final<<<BATCH / 8, 256, 0, stream>>>(h2, W3, b3, out);
}
